// MultiHeadedLatentAttention_11458972745883
// MI455X (gfx1250) — matrix operand order repaired and checked by reading, not compiled, not run on hardware
//
#include <hip/hip_runtime.h>

// ---------------------------------------------------------------------------
// MLA forward for MI455X (gfx1250).  All matmuls via v_wmma_f32_16x16x32_bf16
// with f32 accumulation.  Every WMMA operand is fetched with contiguous
// 16-byte vector accesses:
//   - GEMM A: global_load_b128 (bf16 row-major)
//   - GEMM B: staged per-block in LDS (double-buffered, one global_load_b128
//     per thread per K-step), read back as conflict-free ds_load_b128
//   - attention Q/K/V: global_load_b128 (V stored transposed [b,h,dv,s])
//   - attention P: ds_load_b128 round-trip for layout conversion
// ---------------------------------------------------------------------------

typedef __bf16 bf16_t;
typedef bf16_t v16bf __attribute__((ext_vector_type(16)));
typedef bf16_t v8bf  __attribute__((ext_vector_type(8)));
typedef bf16_t v4bf  __attribute__((ext_vector_type(4)));
typedef float  v8f   __attribute__((ext_vector_type(8)));

#define B_    2
#define S_    2048
#define HID_  2048
#define H_    16
#define DN_   128
#define DR_   64
#define DV_   128
#define R_    512
#define DQ_   192          // DN_ + DR_
#define TOK_  (B_ * S_)    // 4096

#define LDSB_STRIDE 40     // bf16 elems per staged B column (32 + 8 pad, 80 B)

// Build a 16-element bf16 fragment from two contiguous 8-element (16B) runs.
// A order used here (contiguous): lane half h covers K = [h*16, h*16+8) and [h*16+8, +8)
// of the 32-wide K slab -> {p0[0..7], p1[0..7]}.
// B-layout (col per lane): K = half*16 + j -> one contiguous run of 16.
__device__ __forceinline__ v16bf load_frag16(const bf16_t* p0, const bf16_t* p1) {
    v8bf lo = *(const v8bf*)p0;
    v8bf hi = *(const v8bf*)p1;
    v16bf r;
#pragma unroll
    for (int j = 0; j < 8; ++j) { r[j] = lo[j]; r[j + 8] = hi[j]; }
    return r;
}

// ---------------------------------------------------------------------------
// GEMM:  C[M,N] = A[M,K](bf16) * Bt[N,K](bf16, pre-transposed), f32 accum.
// Block = 256 threads = 8 waves; wave w owns rows [blk*128+16w, +16),
// cols [blockIdx.y*64, +64) as 4 WMMA C tiles.  K stepped by 32.
// The 64x32 B tile is staged once per block in double-buffered LDS.
// SPLIT_KV epilogue scatters N=4096 output into k_nope [tok,h,128] and
// V-transposed [b,h,dv,s].
// ---------------------------------------------------------------------------
template <typename TC, bool SPLIT_KV>
__global__ __launch_bounds__(256) void
gemm_bf16_wmma(const bf16_t* __restrict__ A, const bf16_t* __restrict__ Bt,
               TC* __restrict__ C, TC* __restrict__ C2, int M, int N, int K) {
    __shared__ __align__(16) bf16_t Blds[2][64 * LDSB_STRIDE];   // 2 x 5 KB
    const int tid  = threadIdx.x;
    const int lane = tid & 31;
    const int w    = tid >> 5;
    const int half = lane >> 4;
    const int ln16 = lane & 15;
    const int row0 = blockIdx.x * 128 + w * 16;
    const int col0 = blockIdx.y * 64;

    v8f acc[4];
#pragma unroll
    for (int t = 0; t < 4; ++t)
#pragma unroll
        for (int e = 0; e < 8; ++e) acc[t][e] = 0.0f;

    // Cooperative B staging: 4 threads per column, 16 B each per K-step.
    const int lcol = tid >> 2, lchunk = tid & 3;
    const bf16_t* bsrc = Bt + (size_t)(col0 + lcol) * K + lchunk * 8;
    bf16_t* bdst = &Blds[0][0] + lcol * LDSB_STRIDE + lchunk * 8;

    const int arow = row0 + ln16;
    const bf16_t* arp = A + (size_t)arow * K;

    // Preload K-slab 0 into buffer 0.
    v8bf breg = *(const v8bf*)bsrc;
    *(v8bf*)bdst = breg;
    __syncthreads();

    int buf = 0;
    for (int kb = 0; kb < K; kb += 32) {
        const bool has_next = (kb + 32) < K;
        if (has_next) {
            breg = *(const v8bf*)(bsrc + kb + 32);
            __builtin_prefetch(arp + kb + 64, 0, 1);            // global_prefetch_b8
        }
        v16bf af = load_frag16(arp + kb + half * 16, arp + kb + half * 16 + 8);
#pragma unroll
        for (int t = 0; t < 4; ++t) {
            const bf16_t* brp =
                &Blds[buf][0] + (t * 16 + ln16) * LDSB_STRIDE + half * 16;
            v16bf bfv = load_frag16(brp, brp + 8);
            acc[t] = __builtin_amdgcn_wmma_f32_16x16x32_bf16(
                false, af, false, bfv, (short)0, acc[t], false, false);
        }
        if (has_next) {
            // All waves finished reading `buf` (program order above); publish
            // the next slab into the other buffer and make it visible.
            *(v8bf*)(&Blds[buf ^ 1][0] + lcol * LDSB_STRIDE + lchunk * 8) = breg;
            __syncthreads();
            buf ^= 1;
        }
    }
#pragma unroll
    for (int t = 0; t < 4; ++t) {
        const int col = col0 + t * 16 + ln16;
#pragma unroll
        for (int v = 0; v < 8; ++v) {
            const int row = row0 + v + half * 8;
            if (SPLIT_KV) {
                // N = H*256: [0,128) of each head -> k_nope, [128,256) -> vT
                const int h = col >> 8, c = col & 255;
                const int b = row >> 11, s = row & (S_ - 1);
                if (c < DN_)
                    C[(size_t)row * (H_ * DN_) + h * DN_ + c] = (TC)acc[t][v];
                else
                    C2[((size_t)(b * H_ + h) * DV_ + (c - DN_)) * S_ + s] = (TC)acc[t][v];
            } else {
                C[(size_t)row * N + col] = (TC)acc[t][v];
            }
        }
    }
}

// ---------------------------------------------------------------------------
// f32 -> bf16 elementwise convert (hidden states), 4 elems/thread.
// ---------------------------------------------------------------------------
__global__ void cvt_f32_bf16_kernel(const float* __restrict__ src,
                                    bf16_t* __restrict__ dst, int n4) {
    const int i = blockIdx.x * blockDim.x + threadIdx.x;
    if (i >= n4) return;
    const float4 v = *(const float4*)(src + (size_t)i * 4);
    v4bf o; o[0] = (bf16_t)v.x; o[1] = (bf16_t)v.y; o[2] = (bf16_t)v.z; o[3] = (bf16_t)v.w;
    *(v4bf*)(dst + (size_t)i * 4) = o;
}

// ---------------------------------------------------------------------------
// f32 [K,N] -> bf16 [N,K] tiled transpose-convert (weights).  Block (32,8).
// ---------------------------------------------------------------------------
__global__ void transpose_cvt_kernel(const float* __restrict__ src,
                                     bf16_t* __restrict__ dst, int K, int N) {
    __shared__ bf16_t tile[32][33];
    const int n0 = blockIdx.x * 32;
    const int k0 = blockIdx.y * 32;
    const int tx = threadIdx.x, ty = threadIdx.y;
#pragma unroll
    for (int i = 0; i < 32; i += 8)
        tile[ty + i][tx] = (bf16_t)src[(size_t)(k0 + ty + i) * N + n0 + tx];
    __syncthreads();
#pragma unroll
    for (int i = 0; i < 32; i += 8)
        dst[(size_t)(n0 + ty + i) * K + k0 + tx] = tile[tx][ty + i];
}

// ---------------------------------------------------------------------------
// LayerNorm over R=512 of ckv + RoPE on k_pe (last 64).  One block per token.
// ---------------------------------------------------------------------------
__global__ void ln_rope_kernel(const bf16_t* __restrict__ ckv,
                               const float* __restrict__ cosT, const float* __restrict__ sinT,
                               const int* __restrict__ pos_ids,
                               const float* __restrict__ ln_g, const float* __restrict__ ln_b,
                               bf16_t* __restrict__ kvcn, bf16_t* __restrict__ kpe) {
    const int r   = blockIdx.x;
    const int tid = threadIdx.x;       // 128 threads
    const bf16_t* src = ckv + (size_t)r * (R_ + DR_);

    float vals[4], s = 0.f, sq = 0.f;
#pragma unroll
    for (int i = 0; i < 4; ++i) {
        float x = (float)src[tid + i * 128];
        vals[i] = x; s += x; sq += x * x;
    }
    for (int off = 16; off >= 1; off >>= 1) {
        s  += __shfl_xor(s,  off, 32);
        sq += __shfl_xor(sq, off, 32);
    }
    __shared__ float ps[4], pq[4];
    const int wv = tid >> 5;
    if ((tid & 31) == 0) { ps[wv] = s; pq[wv] = sq; }
    __syncthreads();
    s  = ps[0] + ps[1] + ps[2] + ps[3];
    sq = pq[0] + pq[1] + pq[2] + pq[3];
    const float mu  = s / (float)R_;
    const float var = sq / (float)R_ - mu * mu;
    const float inv = rsqrtf(var + 1e-6f);

    bf16_t* dst = kvcn + (size_t)r * R_;
#pragma unroll
    for (int i = 0; i < 4; ++i) {
        const int c = tid + i * 128;
        dst[c] = (bf16_t)((vals[i] - mu) * inv * ln_g[c] + ln_b[c]);
    }
    if (tid < 32) {
        const int p = pos_ids[r];
        const float c = cosT[p * (DR_ / 2) + tid], sn = sinT[p * (DR_ / 2) + tid];
        const float x1 = (float)src[R_ + tid];
        const float x2 = (float)src[R_ + 32 + tid];
        kpe[(size_t)r * DR_ + tid]      = (bf16_t)(x1 * c - x2 * sn);
        kpe[(size_t)r * DR_ + 32 + tid] = (bf16_t)(x2 * c + x1 * sn);
    }
}

// ---------------------------------------------------------------------------
// In-place RoPE on q_pe slice of Q[b,s,h,192] (dims 128..191).
// ---------------------------------------------------------------------------
__global__ void rope_q_kernel(bf16_t* __restrict__ qbf,
                              const float* __restrict__ cosT, const float* __restrict__ sinT,
                              const int* __restrict__ pos_ids, int total) {
    const int idx = blockIdx.x * blockDim.x + threadIdx.x;
    if (idx >= total) return;
    const int i = idx & 31;
    const int h = (idx >> 5) & (H_ - 1);
    const int r = idx >> 9;
    const int p = pos_ids[r];
    const float c = cosT[p * (DR_ / 2) + i], sn = sinT[p * (DR_ / 2) + i];
    bf16_t* base = qbf + (size_t)r * (H_ * DQ_) + h * DQ_ + DN_;
    const float x1 = (float)base[i], x2 = (float)base[32 + i];
    base[i]      = (bf16_t)(x1 * c - x2 * sn);
    base[32 + i] = (bf16_t)(x2 * c + x1 * sn);
}

// ---------------------------------------------------------------------------
// Flash-style causal attention.  Block = 128 threads = 4 waves; each wave one
// 16-row Q tile of one (b,h).  32 keys/iter: 12 score WMMAs (192 features),
// online softmax, P via LDS, 8 WMMAs against V (D_V=128).
// All global fragment loads are contiguous b128; V comes from [b,h,dv,s].
// ---------------------------------------------------------------------------
__global__ __launch_bounds__(128, 1) void
mla_attn_kernel(const bf16_t* __restrict__ qbf, const bf16_t* __restrict__ knope,
                const bf16_t* __restrict__ kpe, const bf16_t* __restrict__ vT,
                bf16_t* __restrict__ aout) {
    __shared__ __align__(16) bf16_t Plds[4][16][32];
    const int lane = threadIdx.x & 31;
    const int wloc = threadIdx.x >> 5;
    const int tile = blockIdx.x * 4 + wloc;
    const int qt = tile & (S_ / 16 - 1);
    const int bh = tile >> 7;
    const int h  = bh & (H_ - 1);
    const int b  = bh >> 4;
    const int q0 = qt * 16;
    const int half = lane >> 4;
    const int ln16 = lane & 15;
    const float scale = 0.07216878364870323f;   // 1/sqrt(192)

    // Q fragments (6 feature blocks of 32) cached in registers.
    v16bf qf[6];
    {
        const bf16_t* qb = qbf + (size_t)(b * S_ + q0 + ln16) * (H_ * DQ_) + h * DQ_;
#pragma unroll
        for (int f = 0; f < 6; ++f)
            qf[f] = load_frag16(qb + f * 32 + half * 16, qb + f * 32 + half * 16 + 8);
    }

    v8f o[8];
#pragma unroll
    for (int t = 0; t < 8; ++t)
#pragma unroll
        for (int e = 0; e < 8; ++e) o[t][e] = 0.0f;
    float mrow[8], lrow[8];
#pragma unroll
    for (int v = 0; v < 8; ++v) { mrow[v] = -1e30f; lrow[v] = 0.0f; }

    const int nkb = (q0 + 16 + 31) >> 5;
    for (int kb = 0; kb < nkb; ++kb) {
        const int key0 = kb * 32;

        // --- scores: two 16x16 fragments (keys 0-15 / 16-31 of block) ---
        v8f sacc[2];
#pragma unroll
        for (int sub = 0; sub < 2; ++sub) {
#pragma unroll
            for (int e = 0; e < 8; ++e) sacc[sub][e] = 0.0f;
            const int krow = b * S_ + key0 + sub * 16 + ln16;
            const bf16_t* knb  = knope + (size_t)krow * (H_ * DN_) + h * DN_;
            const bf16_t* kpeb = kpe + (size_t)krow * DR_;
#pragma unroll
            for (int f = 0; f < 6; ++f) {
                const bf16_t* src = (f < 4) ? (knb + f * 32 + half * 16)
                                            : (kpeb + (f - 4) * 32 + half * 16);
                v16bf bfr = load_frag16(src, src + 8);
                sacc[sub] = __builtin_amdgcn_wmma_f32_16x16x32_bf16(
                    false, qf[f], false, bfr, (short)0, sacc[sub], false, false);
            }
        }

        // --- causal mask + online softmax; P straight into LDS ---
        float alpha[8];
#pragma unroll
        for (int v = 0; v < 8; ++v) {
            const int row = q0 + v + half * 8;
            float s0 = (key0 + ln16 <= row)      ? sacc[0][v] * scale : -1e30f;
            float s1 = (key0 + 16 + ln16 <= row) ? sacc[1][v] * scale : -1e30f;
            float mx = fmaxf(s0, s1);
            for (int off = 1; off < 16; off <<= 1)
                mx = fmaxf(mx, __shfl_xor(mx, off, 32));
            const float mn = fmaxf(mrow[v], mx);
            const float e0 = __expf(s0 - mn);
            const float e1 = __expf(s1 - mn);
            float rs = e0 + e1;
            for (int off = 1; off < 16; off <<= 1)
                rs += __shfl_xor(rs, off, 32);
            const float al = __expf(mrow[v] - mn);
            mrow[v] = mn;
            lrow[v] = lrow[v] * al + rs;
            alpha[v] = al;
            Plds[wloc][v + half * 8][ln16]      = (bf16_t)e0;
            Plds[wloc][v + half * 8][16 + ln16] = (bf16_t)e1;
        }
#pragma unroll
        for (int t = 0; t < 8; ++t)
#pragma unroll
            for (int v = 0; v < 8; ++v) o[t][v] *= alpha[v];

        // --- P as A-fragment: two ds_load_b128 per lane ---
        v16bf pf = load_frag16(&Plds[wloc][ln16][half * 16],
                               &Plds[wloc][ln16][half * 16 + 8]);

        // --- O += P @ V  (V fragments contiguous along key axis) ---
#pragma unroll
        for (int t = 0; t < 8; ++t) {
            const bf16_t* vp = vT + ((size_t)(b * H_ + h) * DV_ + t * 16 + ln16) * S_
                                  + key0 + half * 16;
            v16bf vf = load_frag16(vp, vp + 8);
            o[t] = __builtin_amdgcn_wmma_f32_16x16x32_bf16(
                false, pf, false, vf, (short)0, o[t], false, false);
        }
    }

    // --- epilogue: normalize by row sums, write bf16 attn-out ---
#pragma unroll
    for (int t = 0; t < 8; ++t) {
        const int col = t * 16 + ln16;
#pragma unroll
        for (int v = 0; v < 8; ++v) {
            const int row = b * S_ + q0 + v + half * 8;
            aout[(size_t)row * (H_ * DV_) + h * DV_ + col] =
                (bf16_t)(o[t][v] / lrow[v]);
        }
    }
}

// ---------------------------------------------------------------------------
// Host-side launch.  Input order: hidden, cos, sin, position_ids, Wq, Wkv_a,
// ln_g, ln_b, Wkv_b, Wo.  Output: f32 [B,S,HID].
// ---------------------------------------------------------------------------
extern "C" void kernel_launch(void* const* d_in, const int* in_sizes, int n_in,
                              void* d_out, int out_size, void* d_ws, size_t ws_size,
                              hipStream_t stream) {
    (void)in_sizes; (void)n_in; (void)out_size; (void)ws_size;
    const float* hidden = (const float*)d_in[0];
    const float* cosT   = (const float*)d_in[1];
    const float* sinT   = (const float*)d_in[2];
    const int*   pos    = (const int*)d_in[3];
    const float* Wq     = (const float*)d_in[4];
    const float* Wkv_a  = (const float*)d_in[5];
    const float* ln_g   = (const float*)d_in[6];
    const float* ln_b   = (const float*)d_in[7];
    const float* Wkv_b  = (const float*)d_in[8];
    const float* Wo     = (const float*)d_in[9];
    float* out = (float*)d_out;

    // Workspace layout (bf16 intermediates + transposed bf16 weights), ~124 MB.
    char* ws = (char*)d_ws;
    bf16_t* qbf   = (bf16_t*)(ws);                 // [TOK, H*192]
    bf16_t* ckv   = (bf16_t*)(ws + 25165824);      // [TOK, 576]
    bf16_t* kvcn  = (bf16_t*)(ws + 29884416);      // [TOK, 512]
    bf16_t* kpe   = (bf16_t*)(ws + 34078720);      // [TOK, 64]
    bf16_t* knope = (bf16_t*)(ws + 34603008);      // [TOK, H*128]
    bf16_t* vT    = (bf16_t*)(ws + 51380224);      // [B,H,128,S]
    bf16_t* aout  = (bf16_t*)(ws + 68157440);      // [TOK, H*128]
    bf16_t* hidbf = (bf16_t*)(ws + 84934656);      // [TOK, 2048]
    bf16_t* WqT   = (bf16_t*)(ws + 101711872);     // [3072, 2048]
    bf16_t* WkvaT = (bf16_t*)(ws + 114294784);     // [576, 2048]
    bf16_t* WkvbT = (bf16_t*)(ws + 116654080);     // [4096, 512]
    bf16_t* WoT   = (bf16_t*)(ws + 120848384);     // [2048, 2048]

    // 0) precision/layout conversion
    cvt_f32_bf16_kernel<<<(TOK_ * HID_ / 4 + 255) / 256, 256, 0, stream>>>(
        hidden, hidbf, TOK_ * HID_ / 4);
    transpose_cvt_kernel<<<dim3(3072 / 32, 2048 / 32), dim3(32, 8), 0, stream>>>(
        Wq, WqT, HID_, H_ * DQ_);
    transpose_cvt_kernel<<<dim3(576 / 32, 2048 / 32), dim3(32, 8), 0, stream>>>(
        Wkv_a, WkvaT, HID_, R_ + DR_);
    transpose_cvt_kernel<<<dim3(4096 / 32, 512 / 32), dim3(32, 8), 0, stream>>>(
        Wkv_b, WkvbT, R_, H_ * (DN_ + DV_));
    transpose_cvt_kernel<<<dim3(2048 / 32, 2048 / 32), dim3(32, 8), 0, stream>>>(
        Wo, WoT, H_ * DV_, HID_);

    dim3 blk(256);
    // 1) Q = hidden @ Wq                (4096 x 3072 x 2048)
    gemm_bf16_wmma<bf16_t, false><<<dim3(TOK_ / 128, 3072 / 64), blk, 0, stream>>>(
        hidbf, WqT, qbf, (bf16_t*)nullptr, TOK_, H_ * DQ_, HID_);
    // 2) ckv = hidden @ Wkv_a           (4096 x 576 x 2048)
    gemm_bf16_wmma<bf16_t, false><<<dim3(TOK_ / 128, 576 / 64), blk, 0, stream>>>(
        hidbf, WkvaT, ckv, (bf16_t*)nullptr, TOK_, R_ + DR_, HID_);
    // 3) LayerNorm(kv_c) + RoPE(k_pe)
    ln_rope_kernel<<<TOK_, 128, 0, stream>>>(ckv, cosT, sinT, pos, ln_g, ln_b, kvcn, kpe);
    // 4) RoPE(q_pe) in place
    {
        const int total = TOK_ * H_ * 32;
        rope_q_kernel<<<(total + 255) / 256, 256, 0, stream>>>(qbf, cosT, sinT, pos, total);
    }
    // 5) kv = kv_c_n @ Wkv_b -> k_nope + V-transposed   (4096 x 4096 x 512)
    gemm_bf16_wmma<bf16_t, true><<<dim3(TOK_ / 128, 4096 / 64), blk, 0, stream>>>(
        kvcn, WkvbT, knope, vT, TOK_, H_ * (DN_ + DV_), R_);
    // 6) causal flash attention
    mla_attn_kernel<<<B_ * H_ * (S_ / 16) / 4, 128, 0, stream>>>(
        qbf, knope, kpe, vT, aout);
    // 7) out = attn_out @ Wo            (4096 x 2048 x 2048), f32 output
    gemm_bf16_wmma<float, false><<<dim3(TOK_ / 128, 2048 / 64), blk, 0, stream>>>(
        aout, WoT, out, (float*)nullptr, TOK_, HID_, H_ * DV_);
}
